// SurpriseBasedMemory_51376398795450
// MI455X (gfx1250) — matrix operand order repaired and checked by reading, not compiled, not run on hardware
//
#include <hip/hip_runtime.h>
#include <hip/hip_bf16.h>
#include <math.h>

// Problem sizes (fixed by reference)
#define B_N  65536
#define M_N  2048
#define KD_N 128
#define VD_N 128

typedef __attribute__((ext_vector_type(16))) __bf16          v16bf;
typedef __attribute__((ext_vector_type(16))) unsigned short  v16u;
typedef __attribute__((ext_vector_type(8)))  float           v8f;

__device__ __forceinline__ unsigned short f2bf(float f) {
  unsigned u = __float_as_uint(f);
  u += 0x7FFFu + ((u >> 16) & 1u);            // round-to-nearest-even
  return (unsigned short)(u >> 16);
}

// LDS byte offset of a generic pointer into shared memory (addrspace(3) cast)
__device__ __forceinline__ unsigned lds_off(const void* p) {
  return (unsigned)(unsigned long long)(__attribute__((address_space(3))) const char*)p;
}

// Async copy of a contiguous 16KB block (global -> LDS), 256 threads,
// 4 x B128 per lane.  Tracked by ASYNCcnt.
__device__ __forceinline__ void async_copy_16k(unsigned ldsDst,
                                               const unsigned short* gsrc,
                                               int tid) {
  unsigned off = (unsigned)tid * 16u;
#pragma unroll
  for (int k = 0; k < 4; ++k) {
    asm volatile("global_load_async_to_lds_b128 %0, %1, %2"
                 :: "v"(ldsDst + off + (unsigned)k * 4096u),
                    "v"(off + (unsigned)k * 4096u),
                    "s"(gsrc)
                 : "memory");
  }
}

__device__ __forceinline__ void wait_async0() {
  asm volatile("s_wait_asynccnt 0x0" ::: "memory");
}

// A fragment, 16x32 bf16, ISA layout: lanes0-15 row M=r hold K=0..7,16..23;
// lanes16-31 hold K=8..15,24..31. Source is row-major [row][k], ldr in elems.
__device__ __forceinline__ v16bf loadA(const unsigned short* base, int ldr, int lane) {
  const int half = (lane >> 4) & 1, r = lane & 15;
  const unsigned short* p = base + r * ldr + half * 8;
  v16u t;
#pragma unroll
  for (int e = 0; e < 8; ++e) t[e] = p[e];
#pragma unroll
  for (int e = 0; e < 8; ++e) t[e + 8] = p[e + 16];
  return __builtin_bit_cast(v16bf, t);
}

// B fragment, 32x16 bf16: lanes0-15 K=0..7,16..23, lanes16-31 K=8..15,24..31, col n=lane&15.
// Source stored as [n][k] row-major (ldr elems), i.e. B[k][n] = src[n*ldr+k].
__device__ __forceinline__ v16bf loadB(const unsigned short* baseNK, int ldr, int lane) {
  const int half = (lane >> 4) & 1, n = lane & 15;
  const unsigned short* p = baseNK + n * ldr + half * 8;
  v16u t;
#pragma unroll
  for (int e = 0; e < 8; ++e) t[e] = p[e];
#pragma unroll
  for (int e = 0; e < 8; ++e) t[e + 8] = p[e + 16];
  return __builtin_bit_cast(v16bf, t);
}

__device__ __forceinline__ v8f splat8(float x) {
  v8f r;
#pragma unroll
  for (int e = 0; e < 8; ++e) r[e] = x;
  return r;
}

#define WMMA_BF16(a, b, c) \
  __builtin_amdgcn_wmma_f32_16x16x32_bf16(false, (a), false, (b), (short)0, (c), false, false)

// ---------------------------------------------------------------------------
// prep 1: mem_values -> bf16 chunk-transposed tiles vtg[m/64][c][m%64];
//         Wq, Wv -> bf16
__global__ __launch_bounds__(256) void conv_kernel(
    const float* __restrict__ mv, const float* __restrict__ wq,
    const float* __restrict__ wv, unsigned short* __restrict__ vtg,
    unsigned short* __restrict__ wqbf, unsigned short* __restrict__ wvbf) {
  int i = blockIdx.x * 256 + threadIdx.x;
  if (i < M_N * VD_N) {
    int m = i >> 7, c = i & 127;
    vtg[(m >> 6) * 8192 + c * 64 + (m & 63)] = f2bf(mv[i]);
  }
  if (i < KD_N * KD_N) { wqbf[i] = f2bf(wq[i]); wvbf[i] = f2bf(wv[i]); }
}

// prep 2: kproj[m][j] = dot(mem_keys[m], Wk[j]) + bk[j]  -> bf16
__global__ __launch_bounds__(128) void kproj_kernel(
    const float* __restrict__ mk, const float* __restrict__ Wk,
    const float* __restrict__ bk, unsigned short* __restrict__ kbf) {
  __shared__ float krow[KD_N];
  int m = blockIdx.x, j = threadIdx.x;
  krow[j] = mk[m * KD_N + j];
  __syncthreads();
  float acc = bk[j];
#pragma unroll 4
  for (int c = 0; c < KD_N; ++c) acc += krow[c] * Wk[j * KD_N + c];
  kbf[m * KD_N + j] = f2bf(acc);
}

// prep 3: sw = softmax(surprise_scores) over M
__global__ __launch_bounds__(256) void sw_kernel(const float* __restrict__ ss,
                                                 float* __restrict__ sw) {
  __shared__ float red[256];
  int t = threadIdx.x;
  float m = -1e30f;
  for (int i = t; i < M_N; i += 256) m = fmaxf(m, ss[i]);
  red[t] = m; __syncthreads();
  for (int o = 128; o; o >>= 1) { if (t < o) red[t] = fmaxf(red[t], red[t + o]); __syncthreads(); }
  float mx = red[0]; __syncthreads();
  float s = 0.f;
  for (int i = t; i < M_N; i += 256) s += __expf(ss[i] - mx);
  red[t] = s; __syncthreads();
  for (int o = 128; o; o >>= 1) { if (t < o) red[t] += red[t + o]; __syncthreads(); }
  float Z = red[0];
  for (int i = t; i < M_N; i += 256) sw[i] = __expf(ss[i] - mx) / Z;
}

// ---------------------------------------------------------------------------
// main fused flash kernel: 128 rows/block, 8 waves x 16 rows, all GEMMs via
// WMMA; K/V chunks double-buffered via GLOBAL_LOAD_ASYNC_TO_LDS_B128.
__global__ __launch_bounds__(256) void flash_kernel(
    const float* __restrict__ query, const float* __restrict__ target,
    const unsigned short* __restrict__ kbf, const unsigned short* __restrict__ vtg,
    const unsigned short* __restrict__ wqbf, const unsigned short* __restrict__ wvbf,
    const float* __restrict__ swg, const float* __restrict__ bq,
    const float* __restrict__ bv, float* __restrict__ outRet,
    float* __restrict__ outSup) {
  extern __shared__ char smem[];
  unsigned short* qA    = (unsigned short*)smem;          // 128x128 bf16 (32KB)
  unsigned short* kbuf0 = qA + 128 * 128;                 // 64x128 (16KB)
  unsigned short* vbuf0 = kbuf0 + 64 * 128;               // 128x64 (16KB)
  unsigned short* kbuf1 = vbuf0 + 64 * 128;               // 64x128 (16KB)
  unsigned short* vbuf1 = kbuf1 + 64 * 128;               // 128x64 (16KB)
  float*          swl   = (float*)(vbuf1 + 64 * 128);     // 2048 f32 (8KB)
  unsigned short* Pst   = (unsigned short*)(swl + M_N);   // 8x16x64 (16KB)
  unsigned short* qstage = kbuf1;                         // prologue reuse (32KB)
  unsigned short* wvl    = kbuf0;                         // epilogue reuse (32KB)

  const int tid = threadIdx.x, lane = tid & 31, w = tid >> 5;
  const int half = lane >> 4, nn = lane & 15;
  const int rowBase = blockIdx.x * 128;
  unsigned short* Pw = Pst + w * 16 * 64;
  const unsigned ldsBuf0 = lds_off(kbuf0);
  const unsigned ldsBuf1 = lds_off(kbuf1);

  // kick off chunk 0 (K + V) into buffer 0 while the prologue runs
  async_copy_16k(ldsBuf0,         kbf, tid);
  async_copy_16k(ldsBuf0 + 16384, vtg, tid);

  // prefetch target tile (used only in the epilogue)
  for (int i = tid; i < 1024; i += 256)
    __builtin_prefetch(target + (size_t)rowBase * VD_N + i * 16, 0, 1);

  for (int i = tid; i < M_N; i += 256) swl[i] = swg[i];
  // stage query (bf16, into buffer1) and Wq (bf16, into qA)
  for (int i = tid; i < 128 * 128; i += 256) qstage[i] = f2bf(query[rowBase * 128 + i]);
  for (int i = tid; i < 128 * 128; i += 256) qA[i] = wqbf[i];
  __syncthreads();

  // q projection: q = query @ Wq.T + bq   (A = query rows, B[k][j] = Wq[j][k])
  v8f qacc[8];
#pragma unroll
  for (int nt = 0; nt < 8; ++nt) qacc[nt] = splat8(bq[nt * 16 + nn]);
#pragma unroll
  for (int k = 0; k < 4; ++k) {
    v16bf a = loadA(qstage + w * 16 * 128 + k * 32, 128, lane);
#pragma unroll
    for (int nt = 0; nt < 8; ++nt) {
      v16bf b = loadB(qA + nt * 16 * 128 + k * 32, 128, lane);
      qacc[nt] = WMMA_BF16(a, b, qacc[nt]);
    }
  }
  __syncthreads();
#pragma unroll
  for (int nt = 0; nt < 8; ++nt)
#pragma unroll
    for (int rr = 0; rr < 8; ++rr)
      qA[(w * 16 + rr + 8 * half) * 128 + nt * 16 + nn] = f2bf(qacc[nt][rr]);
  wait_async0();            // chunk 0 landed in LDS
  __syncthreads();

  // online-softmax state per wave-row
  float mx[8], As[8], Bd[8];
  v8f O[8];
#pragma unroll
  for (int rr = 0; rr < 8; ++rr) { mx[rr] = -1e30f; As[rr] = 0.f; Bd[rr] = 0.f; }
#pragma unroll
  for (int nt = 0; nt < 8; ++nt) O[nt] = splat8(0.f);
  const float scale = 0.08838834764831845f;  // 1/sqrt(128)

  for (int c = 0; c < M_N / 64; ++c) {
    const int m0 = c * 64;
    const unsigned short* kN = (c & 1) ? kbuf1 : kbuf0;
    const unsigned short* vT = (c & 1) ? vbuf1 : vbuf0;
    const unsigned ldsNext = (c & 1) ? ldsBuf0 : ldsBuf1;
    // prefetch next K/V chunk; on the last iteration pull Wv instead
    const unsigned short* s0 = (c + 1 < M_N / 64) ? kbf + (m0 + 64) * 128 : wvbf;
    const unsigned short* s1 = (c + 1 < M_N / 64) ? vtg + (m0 + 64) * 128 : wvbf + 8192;
    async_copy_16k(ldsNext,         s0, tid);
    async_copy_16k(ldsNext + 16384, s1, tid);

    // S = q @ k.T  (B[k][n] = kproj[m0+n][k])
    v8f sc[4];
#pragma unroll
    for (int t = 0; t < 4; ++t) sc[t] = splat8(0.f);
#pragma unroll
    for (int k = 0; k < 4; ++k) {
      v16bf a = loadA(qA + w * 16 * 128 + k * 32, 128, lane);
#pragma unroll
      for (int t = 0; t < 4; ++t) {
        v16bf b = loadB(kN + t * 16 * 128 + k * 32, 128, lane);
        sc[t] = WMMA_BF16(a, b, sc[t]);
      }
    }

    // online softmax update (rows live per C-layout: VGPR rr, cols across 16 lanes)
    float corr[8];
#pragma unroll
    for (int rr = 0; rr < 8; ++rr) {
      float cm = -1e30f;
#pragma unroll
      for (int t = 0; t < 4; ++t) { sc[t][rr] *= scale; cm = fmaxf(cm, sc[t][rr]); }
#pragma unroll
      for (int off = 1; off < 16; off <<= 1) cm = fmaxf(cm, __shfl_xor(cm, off, 32));
      float nm = fmaxf(mx[rr], cm);
      corr[rr] = __expf(mx[rr] - nm);
      mx[rr] = nm;
      float eS = 0.f, wS = 0.f;
#pragma unroll
      for (int t = 0; t < 4; ++t) {
        float e  = __expf(sc[t][rr] - nm);
        float wv = e * swl[m0 + t * 16 + nn];
        eS += e; wS += wv;
        Pw[(rr + 8 * half) * 64 + t * 16 + nn] = f2bf(wv);
      }
#pragma unroll
      for (int off = 1; off < 16; off <<= 1) {
        eS += __shfl_xor(eS, off, 32);
        wS += __shfl_xor(wS, off, 32);
      }
      As[rr] = As[rr] * corr[rr] + eS;
      Bd[rr] = Bd[rr] * corr[rr] + wS;
    }
#pragma unroll
    for (int nt = 0; nt < 8; ++nt)
#pragma unroll
      for (int rr = 0; rr < 8; ++rr) O[nt][rr] *= corr[rr];

    // O += P @ V   (B[k][n] = v[m0+k][n] staged as vT[n][k])
#pragma unroll
    for (int kk = 0; kk < 2; ++kk) {
      v16bf a = loadA(Pw + kk * 32, 64, lane);
#pragma unroll
      for (int nt = 0; nt < 8; ++nt) {
        v16bf b = loadB(vT + nt * 16 * 64 + kk * 32, 64, lane);
        O[nt] = WMMA_BF16(a, b, O[nt]);
      }
    }
    wait_async0();           // next chunk (or Wv) is in LDS
    __syncthreads();
  }

  // epilogue: retrieved = (O / (Bd + 1e-8*A)) @ Wv.T + bv ;  Wv already in LDS
  float rinv[8];
#pragma unroll
  for (int rr = 0; rr < 8; ++rr) rinv[rr] = 1.f / (Bd[rr] + 1e-8f * As[rr]);
  v8f acc[8];
#pragma unroll
  for (int nt = 0; nt < 8; ++nt) acc[nt] = splat8(bv[nt * 16 + nn]);
#pragma unroll
  for (int kh = 0; kh < 2; ++kh) {
#pragma unroll
    for (int t = 0; t < 4; ++t)
#pragma unroll
      for (int rr = 0; rr < 8; ++rr)
        Pw[(rr + 8 * half) * 64 + t * 16 + nn] = f2bf(O[kh * 4 + t][rr] * rinv[rr]);
#pragma unroll
    for (int kk = 0; kk < 2; ++kk) {
      v16bf a = loadA(Pw + kk * 32, 64, lane);
#pragma unroll
      for (int nt = 0; nt < 8; ++nt) {
        v16bf b = loadB(wvl + nt * 16 * 128 + kh * 64 + kk * 32, 128, lane);
        acc[nt] = WMMA_BF16(a, b, acc[nt]);
      }
    }
  }

  // store retrieved + surprise = ||target - retrieved||
  float sq[8] = {0.f, 0.f, 0.f, 0.f, 0.f, 0.f, 0.f, 0.f};
#pragma unroll
  for (int nt = 0; nt < 8; ++nt) {
    int col = nt * 16 + nn;
#pragma unroll
    for (int rr = 0; rr < 8; ++rr) {
      int row = rowBase + w * 16 + rr + 8 * half;
      float o = acc[nt][rr];
      outRet[row * VD_N + col] = o;
      float e = target[row * VD_N + col] - o;
      sq[rr] += e * e;
    }
  }
#pragma unroll
  for (int rr = 0; rr < 8; ++rr)
#pragma unroll
    for (int off = 1; off < 16; off <<= 1) sq[rr] += __shfl_xor(sq[rr], off, 32);
  if (nn == 0) {
#pragma unroll
    for (int rr = 0; rr < 8; ++rr)
      outSup[rowBase + w * 16 + rr + 8 * half] = sqrtf(sq[rr]);
  }
}

// ---------------------------------------------------------------------------
// sequential replace-min scan (1 wave). ss kept in LDS with a 64-way min-tree;
// only last-writer indices recorded (copies deferred to writeback kernel).
__global__ __launch_bounds__(32) void scan_kernel(const float* __restrict__ ss_in,
                                                  const float* __restrict__ sup,
                                                  float* __restrict__ ns_out,
                                                  int* __restrict__ last_out) {
  __shared__ float ssl[M_N];
  __shared__ int   last[M_N];
  __shared__ float lvl[64];
  __shared__ int   lidx[64];
  const int lane = threadIdx.x;

  for (int i = lane; i < M_N; i += 32) { ssl[i] = ss_in[i]; last[i] = -1; }
  for (int c = lane; c < 64; c += 32) {
    float bv_ = ssl[c * 32]; int bi = c * 32;
    for (int e = 1; e < 32; ++e) {
      float v = ssl[c * 32 + e];
      if (v < bv_) { bv_ = v; bi = c * 32 + e; }
    }
    lvl[c] = bv_; lidx[c] = bi;
  }
  __syncthreads();

  for (int i = 0; i < B_N; ++i) {
    float v0 = lvl[lane];      int i0 = lidx[lane];
    float v1 = lvl[lane + 32]; int i1 = lidx[lane + 32];
    if (v1 < v0 || (v1 == v0 && i1 < i0)) { v0 = v1; i0 = i1; }
#pragma unroll
    for (int off = 16; off; off >>= 1) {
      float ov = __shfl_xor(v0, off, 32);
      int   oi = __shfl_xor(i0, off, 32);
      if (ov < v0 || (ov == v0 && oi < i0)) { v0 = ov; i0 = oi; }
    }
    float si = sup[i];
    if (si > 0.5f && si > v0) {            // uniform across the wave
      if (lane == 0) { ssl[i0] = si; last[i0] = i; }
      int cb = (i0 >> 5) << 5;
      float v = ssl[cb + lane]; int ix = cb + lane;
#pragma unroll
      for (int off = 16; off; off >>= 1) {
        float ov = __shfl_xor(v, off, 32);
        int   oi = __shfl_xor(ix, off, 32);
        if (ov < v || (ov == v && oi < ix)) { v = ov; ix = oi; }
      }
      if (lane == 0) { lvl[i0 >> 5] = v; lidx[i0 >> 5] = ix; }
    }
  }
  for (int m = lane; m < M_N; m += 32) {
    ns_out[m] = ssl[m] * 0.99f;
    last_out[m] = last[m];
  }
}

// writeback: gather final keys/values per slot
__global__ __launch_bounds__(128) void wb_kernel(
    const float* __restrict__ query, const float* __restrict__ target,
    const float* __restrict__ mk, const float* __restrict__ mv,
    const int* __restrict__ last, float* __restrict__ nk, float* __restrict__ nv) {
  int m = blockIdx.x, t = threadIdx.x;
  int li = last[m];
  nk[m * KD_N + t] = (li >= 0) ? query[(size_t)li * KD_N + t] : mk[m * KD_N + t];
  nv[m * VD_N + t] = (li >= 0) ? target[(size_t)li * VD_N + t] : mv[m * VD_N + t];
}

// ---------------------------------------------------------------------------
extern "C" void kernel_launch(void* const* d_in, const int* in_sizes, int n_in,
                              void* d_out, int out_size, void* d_ws, size_t ws_size,
                              hipStream_t stream) {
  const float* query  = (const float*)d_in[0];
  const float* target = (const float*)d_in[1];
  const float* mk     = (const float*)d_in[2];
  const float* mv     = (const float*)d_in[3];
  const float* sscore = (const float*)d_in[4];
  const float* Wq     = (const float*)d_in[5];
  const float* bq     = (const float*)d_in[6];
  const float* Wk     = (const float*)d_in[7];
  const float* bk     = (const float*)d_in[8];
  const float* Wv     = (const float*)d_in[9];
  const float* bv     = (const float*)d_in[10];

  float* out    = (float*)d_out;
  float* outRet = out;                                  // [B, VD]
  float* outSup = outRet + (size_t)B_N * VD_N;          // [B]
  float* outNK  = outSup + B_N;                         // [M, KD]
  float* outNV  = outNK + (size_t)M_N * KD_N;           // [M, VD]
  float* outNS  = outNV + (size_t)M_N * VD_N;           // [M]

  char* w = (char*)d_ws;
  unsigned short* kbf  = (unsigned short*)w;  w += (size_t)M_N * KD_N * 2;
  unsigned short* vtg  = (unsigned short*)w;  w += (size_t)M_N * VD_N * 2;
  unsigned short* wqbf = (unsigned short*)w;  w += (size_t)KD_N * KD_N * 2;
  unsigned short* wvbf = (unsigned short*)w;  w += (size_t)KD_N * KD_N * 2;
  float*          swb  = (float*)w;           w += (size_t)M_N * 4;
  int*            lastb = (int*)w;            w += (size_t)M_N * 4;

  conv_kernel<<<(M_N * VD_N + 255) / 256, 256, 0, stream>>>(mv, Wq, Wv, vtg, wqbf, wvbf);
  kproj_kernel<<<M_N, 128, 0, stream>>>(mk, Wk, bk, kbf);
  sw_kernel<<<1, 256, 0, stream>>>(sscore, swb);

  const size_t smem = (128 * 128 + 2 * (64 * 128 + 128 * 64) + 8 * 16 * 64) * 2 + M_N * 4;
  flash_kernel<<<B_N / 128, 256, smem, stream>>>(query, target, kbf, vtg, wqbf, wvbf,
                                                 swb, bq, bv, outRet, outSup);

  scan_kernel<<<1, 32, 0, stream>>>(sscore, outSup, outNS, lastb);
  wb_kernel<<<M_N, 128, 0, stream>>>(query, target, mk, mv, lastb, outNK, outNV);
}
